// CapsuleSequenceToGraph_21354577396335
// MI455X (gfx1250) — compile-verified, matrix operand order checked by reading, not run on hardware
//
#include <hip/hip_runtime.h>
#include <hip/hip_bf16.h>

#define T_DIM 50
#define BATCH 64
#define NC 16
#define DC 128
#define MULTI_DIM 128

typedef __attribute__((ext_vector_type(16))) __bf16        v16bf;
typedef __attribute__((ext_vector_type(8)))  float         v8f;
typedef __attribute__((ext_vector_type(8)))  unsigned int  v8u;

// Pack two f32 into one dword of bf16 (round-to-nearest-even), low half = first (even K).
__device__ __forceinline__ unsigned pack_bf16(float a, float b) {
  unsigned ua = __builtin_bit_cast(unsigned, a);
  unsigned ub = __builtin_bit_cast(unsigned, b);
  ua = (ua + 0x7FFFu + ((ua >> 16) & 1u)) >> 16;
  ub = (ub + 0x7FFFu + ((ub >> 16) & 1u)) >> 16;
  return (ub << 16) | (ua & 0xFFFFu);
}

// Load one 16-bit A/B^T fragment (16x32) for the current 32-wide K step.
// `row` points at this lane's matrix row (packed bf16 pairs) + kstep*16 dwords.
// Per ISA layout: dword v holds K-pair (v>>2)*16 + half*8 + (v&3)*2.
__device__ __forceinline__ v16bf ld_frag(const unsigned* row, int hl) {
  v8u u;
  const unsigned* p = row + hl * 4;
  u[0] = p[0]; u[1] = p[1]; u[2] = p[2]; u[3] = p[3];
  p = row + 8 + hl * 4;
  u[4] = p[0]; u[5] = p[1]; u[6] = p[2]; u[7] = p[3];
  return __builtin_bit_cast(v16bf, u);
}

__device__ __forceinline__ v8f wmma_bf16(v16bf a, v16bf b, v8f c) {
  return __builtin_amdgcn_wmma_f32_16x16x32_bf16(
      /*neg_a=*/false, a, /*neg_b=*/false, b,
      /*c_mod=*/(short)0, c, /*reuse_a=*/false, /*reuse_b=*/false);
}

// ---------------------------------------------------------------------------
// Kernel 1: primary capsules. One WG per (t, n_cap); GEMM M=64(batch) x N=128(d_c) x K=J.
// caps[b, t, n, d] = sum_j x[t,b,j] * W[t,n,j,d].  HBM-bound on W (read once, f32).
// ---------------------------------------------------------------------------
__global__ __launch_bounds__(256)
void caps_gemm_kernel(const float* __restrict__ x, const float* __restrict__ W,
                      float* __restrict__ caps, int J) {
  __shared__ unsigned Ap[64 * 16];    // 64 rows (batch) x 32 K as packed pairs
  __shared__ unsigned Bp[128 * 16];   // 128 rows (d_c, = B^T) x 32 K packed

  const int t    = blockIdx.x / NC;
  const int ncap = blockIdx.x % NC;
  const int tid  = threadIdx.x;
  const int lane = tid & 31;
  const int wv   = tid >> 5;          // 8 waves
  const int hl   = lane >> 4;
  const int mrow = lane & 15;
  const int mb   = wv & 3;            // M block (4 x 16 = 64 batch rows)
  const int nb2  = wv >> 2;           // N half (2 x 64 cols)

  const float* xb = x + (size_t)t * BATCH * J;
  const float* Wb = W + ((size_t)t * NC + ncap) * (size_t)J * DC;

  v8f acc[4] = {};
  for (int kb = 0; kb < J; kb += 32) {
    if (kb + 32 < J)
      __builtin_prefetch(Wb + (size_t)(kb + 32) * DC, 0, 1);  // global_prefetch next W tile

    // Stage A panel (64x32 f32 -> packed bf16), coalesced.
    for (int s = tid; s < 64 * 16; s += 256) {
      int b = s >> 4, dc = s & 15;
      int j = kb + dc * 2;
      Ap[s] = pack_bf16(xb[b * J + j], xb[b * J + j + 1]);
    }
    // Stage B^T panel: W rows are [j][d]; consecutive lanes read consecutive d (coalesced).
    for (int i = 0; i < 8; ++i) {
      int s = tid + i * 256;
      int n = s & 127;
      int dc = s >> 7;
      int j = kb + dc * 2;
      Bp[n * 16 + dc] = pack_bf16(Wb[(size_t)j * DC + n], Wb[(size_t)(j + 1) * DC + n]);
    }
    __syncthreads();

    v16bf af = ld_frag(Ap + (mb * 16 + mrow) * 16, hl);
#pragma unroll
    for (int jt = 0; jt < 4; ++jt) {
      v16bf bf = ld_frag(Bp + ((nb2 * 4 + jt) * 16 + mrow) * 16, hl);
      acc[jt] = wmma_bf16(af, bf, acc[jt]);
    }
    __syncthreads();
  }

  // Epilogue: D layout -> lane holds n = lane&15, VGPR v holds m = v + 8*half.
#pragma unroll
  for (int jt = 0; jt < 4; ++jt) {
    int d = (nb2 * 4 + jt) * 16 + mrow;
#pragma unroll
    for (int vi = 0; vi < 8; ++vi) {
      int brow = mb * 16 + vi + 8 * hl;
      caps[(((size_t)brow * T_DIM + t) * NC + ncap) * DC + d] = acc[jt][vi];
    }
  }
}

// ---------------------------------------------------------------------------
// Kernel 2: dynamic routing (3 iters + final) -> tanh vertex. One WG per batch b.
// caps passes stay in L2 (78 MB << 192 MB).
// ---------------------------------------------------------------------------
__global__ __launch_bounds__(256)
void routing_kernel(const float* __restrict__ caps, float* __restrict__ vert) {
  __shared__ float lg[T_DIM * NC];
  __shared__ float rc[T_DIM * NC];
  __shared__ float vsh[NC * DC];
  const int b    = blockIdx.x;
  const int tid  = threadIdx.x;
  const int wv   = tid >> 5;
  const int lane = tid & 31;
  const float* cb = caps + (size_t)b * T_DIM * NC * DC;

  for (int s = tid; s < T_DIM * NC; s += 256) lg[s] = 0.f;
  __syncthreads();

  for (int it = 0; it <= 3; ++it) {
    // softmax over n per t (50 threads active)
    if (tid < T_DIM) {
      float mx = -1e30f;
      for (int n = 0; n < NC; ++n) mx = fmaxf(mx, lg[tid * NC + n]);
      float e[NC], sm = 0.f;
      for (int n = 0; n < NC; ++n) { e[n] = __expf(lg[tid * NC + n] - mx); sm += e[n]; }
      float inv = 1.f / sm;
      for (int n = 0; n < NC; ++n) rc[tid * NC + n] = e[n] * inv;
    }
    __syncthreads();

    // v[n][d] = tanh(sum_t caps[t][n][d] * rc[t][n])
    for (int s = tid; s < NC * DC; s += 256) {
      int n = s >> 7, d = s & 127;
      float a = 0.f;
      for (int t = 0; t < T_DIM; ++t)
        a += cb[((size_t)t * NC + n) * DC + d] * rc[t * NC + n];
      vsh[s] = tanhf(a);
    }
    __syncthreads();

    if (it < 3) {
      // lg[t][n] += dot(v[n], caps[t][n]); wave32 shuffle reduction
      for (int p = wv; p < T_DIM * NC; p += 8) {
        int t = p >> 4, n = p & 15;
        float s = 0.f;
        for (int d = lane; d < DC; d += 32)
          s += vsh[n * DC + d] * cb[((size_t)t * NC + n) * DC + d];
        s += __shfl_xor(s, 16, 32);
        s += __shfl_xor(s, 8, 32);
        s += __shfl_xor(s, 4, 32);
        s += __shfl_xor(s, 2, 32);
        s += __shfl_xor(s, 1, 32);
        if (lane == 0) lg[p] += s;
      }
      __syncthreads();
    }
  }
  for (int s = tid; s < NC * DC; s += 256)
    vert[(size_t)b * NC * DC + s] = vsh[s];
}

// ---------------------------------------------------------------------------
// Kernel 3: fc layer -> multi_vertex rows. One WG per b; M=16, N=128, K=128 via WMMA.
// fc_w is [MULTI][D_C] row-major == B^T layout directly.
// ---------------------------------------------------------------------------
__global__ __launch_bounds__(256)
void fc_kernel(const float* __restrict__ vert, const float* __restrict__ fw,
               const float* __restrict__ fb, float* __restrict__ mv, int modrow) {
  __shared__ unsigned Ap[16 * 64];
  __shared__ unsigned Bp[128 * 64];
  const int b = blockIdx.x, tid = threadIdx.x;
  const int wv = tid >> 5, lane = tid & 31;
  const int hl = lane >> 4, mrow = lane & 15;
  const float* vb = vert + (size_t)b * NC * DC;

  for (int s = tid; s < 16 * 64; s += 256) {
    int r = s >> 6, dc = (s & 63) * 2;
    Ap[s] = pack_bf16(vb[r * DC + dc], vb[r * DC + dc + 1]);
  }
  for (int s = tid; s < 128 * 64; s += 256) {
    int o = s >> 6, dc = (s & 63) * 2;
    Bp[s] = pack_bf16(fw[o * DC + dc], fw[o * DC + dc + 1]);
  }
  __syncthreads();

  v8f acc = {};
#pragma unroll
  for (int ks = 0; ks < 4; ++ks) {
    v16bf af = ld_frag(Ap + mrow * 64 + ks * 16, hl);
    v16bf bf = ld_frag(Bp + (wv * 16 + mrow) * 64 + ks * 16, hl);
    acc = wmma_bf16(af, bf, acc);
  }
  int o = wv * 16 + mrow;
  float bias = fb[o];
#pragma unroll
  for (int vi = 0; vi < 8; ++vi) {
    int m = vi + 8 * hl;
    mv[((size_t)b * 48 + modrow + m) * MULTI_DIM + o] = acc[vi] + bias;
  }
}

// ---------------------------------------------------------------------------
// Kernel 4: adjacency = I + relu(Q K^T / 128). One WG per b. Q/K scalar (tiny),
// Q K^T via WMMA (K rows row-major == B^T layout). n = 16 or 48.
// ---------------------------------------------------------------------------
__global__ __launch_bounds__(256)
void adj_kernel(const float* __restrict__ vert, int n,
                const float* __restrict__ WQ, const float* __restrict__ WK,
                float* __restrict__ adj) {
  __shared__ float    Vf[48 * DC];
  __shared__ unsigned Qp[48 * 64];
  __shared__ unsigned Kp[48 * 64];
  const int b = blockIdx.x, tid = threadIdx.x;
  const int wv = tid >> 5, lane = tid & 31;
  const int hl = lane >> 4, mrow = lane & 15;
  const float* vb = vert + (size_t)b * n * DC;

  for (int s = tid; s < n * DC; s += 256) Vf[s] = vb[s];
  __syncthreads();

  for (int p = tid; p < n * 64; p += 256) {
    int i = p >> 6, oc = (p & 63) * 2;
    float q0 = 0, q1 = 0, k0 = 0, k1 = 0;
    for (int d = 0; d < DC; ++d) {
      float vv = Vf[i * DC + d];
      q0 += vv * WQ[d * DC + oc];
      q1 += vv * WQ[d * DC + oc + 1];
      k0 += vv * WK[d * DC + oc];
      k1 += vv * WK[d * DC + oc + 1];
    }
    Qp[p] = pack_bf16(q0, q1);
    Kp[p] = pack_bf16(k0, k1);
  }
  __syncthreads();

  const int nt = n >> 4;
  for (int tt = wv; tt < nt * nt; tt += 8) {
    int ti = tt / nt, tj = tt % nt;
    v8f acc = {};
#pragma unroll
    for (int ks = 0; ks < 4; ++ks) {
      v16bf af = ld_frag(Qp + (ti * 16 + mrow) * 64 + ks * 16, hl);
      v16bf bf = ld_frag(Kp + (tj * 16 + mrow) * 64 + ks * 16, hl);
      acc = wmma_bf16(af, bf, acc);
    }
    int col = tj * 16 + mrow;
#pragma unroll
    for (int vi = 0; vi < 8; ++vi) {
      int row = ti * 16 + vi + 8 * hl;
      float r = fmaxf(acc[vi] * (1.0f / 128.0f), 0.0f) + (row == col ? 1.0f : 0.0f);
      adj[((size_t)b * n + row) * n + col] = r;
    }
  }
}

// ---------------------------------------------------------------------------
extern "C" void kernel_launch(void* const* d_in, const int* in_sizes, int n_in,
                              void* d_out, int out_size, void* d_ws, size_t ws_size,
                              hipStream_t stream) {
  (void)in_sizes; (void)n_in; (void)out_size; (void)ws_size;
  const float* text  = (const float*)d_in[0];
  const float* audio = (const float*)d_in[1];
  const float* video = (const float*)d_in[2];
  // d_in[3] = batch_size scalar (unused; compile-time B)
  const float* W_tpc = (const float*)d_in[4];
  const float* W_apc = (const float*)d_in[5];
  const float* W_vpc = (const float*)d_in[6];
  const float* WQt = (const float*)d_in[7];
  const float* WKt = (const float*)d_in[8];
  const float* WQa = (const float*)d_in[9];
  const float* WKa = (const float*)d_in[10];
  const float* WQv = (const float*)d_in[11];
  const float* WKv = (const float*)d_in[12];
  const float* WQm = (const float*)d_in[13];
  const float* WKm = (const float*)d_in[14];
  const float* fct_w = (const float*)d_in[15];
  const float* fct_b = (const float*)d_in[16];
  const float* fca_w = (const float*)d_in[17];
  const float* fca_b = (const float*)d_in[18];
  const float* fcv_w = (const float*)d_in[19];
  const float* fcv_b = (const float*)d_in[20];

  float* out = (float*)d_out;
  float* ws  = (float*)d_ws;

  const size_t CAPS = (size_t)BATCH * T_DIM * NC * DC;  // 6,553,600 floats each
  float* t_caps = ws;
  float* a_caps = ws + CAPS;
  float* v_caps = ws + 2 * CAPS;

  // Flat output offsets (floats), reference return order.
  const size_t VT = 0, VA = 131072, VV = 262144, MV = 393216;
  const size_t AT = 786432, AA = 802816, AV = 819200, AM = 835584;

  dim3 gc(T_DIM * NC);  // 800 workgroups per modality
  caps_gemm_kernel<<<gc, 256, 0, stream>>>(text,  W_tpc, t_caps, 768);
  caps_gemm_kernel<<<gc, 256, 0, stream>>>(audio, W_apc, a_caps, 256);
  caps_gemm_kernel<<<gc, 256, 0, stream>>>(video, W_vpc, v_caps, 512);

  routing_kernel<<<BATCH, 256, 0, stream>>>(t_caps, out + VT);
  routing_kernel<<<BATCH, 256, 0, stream>>>(a_caps, out + VA);
  routing_kernel<<<BATCH, 256, 0, stream>>>(v_caps, out + VV);

  fc_kernel<<<BATCH, 256, 0, stream>>>(out + VT, fct_w, fct_b, out + MV, 0);
  fc_kernel<<<BATCH, 256, 0, stream>>>(out + VA, fca_w, fca_b, out + MV, 16);
  fc_kernel<<<BATCH, 256, 0, stream>>>(out + VV, fcv_w, fcv_b, out + MV, 32);

  adj_kernel<<<BATCH, 256, 0, stream>>>(out + VT, 16, WQt, WKt, out + AT);
  adj_kernel<<<BATCH, 256, 0, stream>>>(out + VA, 16, WQa, WKa, out + AA);
  adj_kernel<<<BATCH, 256, 0, stream>>>(out + VV, 16, WQv, WKv, out + AV);
  adj_kernel<<<BATCH, 256, 0, stream>>>(out + MV, 48, WQm, WKm, out + AM);
}
